// LSTM_FC_10668698763529
// MI455X (gfx1250) — matrix operand order repaired and checked by reading, not compiled, not run on hardware
//
#include <hip/hip_runtime.h>

typedef _Float16 half_t;
typedef __attribute__((ext_vector_type(16))) _Float16 v16h;
typedef __attribute__((ext_vector_type(8)))  _Float16 v8h;
typedef __attribute__((ext_vector_type(8)))  float    v8f;

union V16 { v16h v; v8h h[2]; };
union Q8  { v8h v; unsigned long long q[2]; };

__device__ __forceinline__ v8f wmma_f16(v16h a, v16h b, v8f c) {
  return __builtin_amdgcn_wmma_f32_16x16x32_f16(false, a, false, b, (short)0, c, false, false);
}

__device__ __forceinline__ float fsigm(float v) { return 1.0f / (1.0f + __expf(-v)); }
// tanh(x) = 1 - 2/(exp(2x)+1); __expf saturates so +/-inf behave correctly.
__device__ __forceinline__ float ftanh(float v) { return 1.0f - 2.0f / (__expf(2.0f * v) + 1.0f); }

// Opaque 0 (one SALU): defeats LICM so per-step weight-tile loads stay inside the
// timestep loop (L2-resident shared reads) instead of being hoisted and spilled
// to per-wave scratch under the occupancy cap.
__device__ __forceinline__ int opaque_zero() {
  int v;
  asm volatile("s_mov_b32 %0, 0" : "=s"(v));
  return v;
}

// ---------------- weight prepack kernels ----------------
// Repack [4*68][K] fp32 (PyTorch gate order i,f,g,o) into [320][Kpad] fp16,
// gate g -> rows g*80 .. g*80+67, rows 68..79 and cols >= srcK zeroed.
__global__ void pack_gates_kernel(half_t* __restrict__ dst, const float* __restrict__ src,
                                  int srcK, int dstK) {
  int idx = blockIdx.x * blockDim.x + threadIdx.x;
  int total = 320 * dstK;
  if (idx >= total) return;
  int r = idx / dstK, k = idx - r * dstK;
  int g = r / 80, j = r - g * 80;
  float v = (j < 68 && k < srcK) ? src[(g * 68 + j) * srcK + k] : 0.0f;
  dst[idx] = (half_t)v;
}

__global__ void pack_bias_kernel(float* __restrict__ dst, const float* __restrict__ bih,
                                 const float* __restrict__ bhh) {
  int idx = blockIdx.x * blockDim.x + threadIdx.x;
  if (idx >= 320) return;
  int g = idx / 80, j = idx - g * 80;
  dst[idx] = (j < 68) ? (bih[g * 68 + j] + bhh[g * 68 + j]) : 0.0f;
}

// fc_w (5,6528) -> fp16 [16][48][2][96]; n>=5 or c>=68 zero padded.
__global__ void pack_fc_kernel(half_t* __restrict__ dst, const float* __restrict__ fcw) {
  int idx = blockIdx.x * blockDim.x + threadIdx.x;
  const int total = 16 * 48 * 2 * 96;
  if (idx >= total) return;
  int n = idx / 9216;
  int rem = idx - n * 9216;
  int t = rem / 192;
  int rem2 = rem - t * 192;
  int d = rem2 / 96;
  int c = rem2 - d * 96;
  float v = (n < 5 && c < 68) ? fcw[n * 6528 + t * 136 + d * 68 + c] : 0.0f;
  dst[idx] = (half_t)v;
}

__global__ void fc_final_kernel(float* __restrict__ out, const float* __restrict__ ypart,
                                const float* __restrict__ fcb, int total) {
  int idx = blockIdx.x * blockDim.x + threadIdx.x;
  if (idx >= total) return;
  int n = idx % 5;
  out[idx] = ypart[idx] + ypart[total + idx] + fcb[n];
}

// ---------------- fused bidirectional LSTM layer ----------------
// grid = (256, 2): blockIdx.y = direction (0 fwd, 1 bwd). 128 threads = 4 waves,
// each wave owns a 16-row batch tile and runs the full T=48 recurrence in WMMA.
// min 2 waves/SIMD so one wave's elementwise recurrence tail overlaps the other
// wave's WMMA phase; opaque_zero() keeps weight loads in-loop (no spill).
template <int LAYER>
__global__ __launch_bounds__(128, 2) void lstm_kernel(
    const float*  __restrict__ x,     // LAYER 0: [B][48][34] fp32
    const half_t* __restrict__ hin,   // LAYER 1: [B][48][136] fp16
    const half_t* __restrict__ Wih,   // [2][320][KIN] fp16
    const half_t* __restrict__ Whh,   // [2][320][96]  fp16
    const float*  __restrict__ bias,  // [2][320] fp32 (b_ih+b_hh)
    half_t*       __restrict__ hout,  // LAYER 0: [B][48][136] fp16
    const half_t* __restrict__ fcw,   // LAYER 1: [16][48][2][96] fp16
    float*        __restrict__ ypart) // LAYER 1: [2][B][5] fp32
{
  constexpr int KIN = (LAYER == 0) ? 64 : 160;
  constexpr int KC  = (LAYER == 0) ? 2 : 5;

  const int dir  = blockIdx.y;
  const bool rev = (dir == 1);
  const int tid  = threadIdx.x;
  const int wave = tid >> 5;
  const int lane = tid & 31;
  const int m    = lane & 15;
  const bool hi  = lane >= 16;
  const int b0   = (blockIdx.x * 4 + wave) * 16;

  extern __shared__ char smem[];
  half_t* sWhh = (half_t*)smem;                             // [320][96]
  half_t* hT   = (half_t*)(smem + 320 * 96 * 2) + wave * (16 * 96); // per-wave h tile

  // cooperative load of this direction's recurrent weights into LDS
  {
    const unsigned long long* s = (const unsigned long long*)(Whh + (size_t)dir * 320 * 96);
    unsigned long long* d = (unsigned long long*)sWhh;
    for (int i = tid; i < 320 * 96 / 4; i += 128) d[i] = s[i];
  }
  for (int i = lane; i < 16 * 96; i += 32) hT[i] = (half_t)0.0f;
  __syncthreads();

  const half_t* Wih_d  = Wih + (size_t)dir * 320 * KIN;
  const float*  bias_d = bias + dir * 320;

  float breg[4][5];
#pragma unroll
  for (int g = 0; g < 4; ++g)
#pragma unroll
    for (int s = 0; s < 5; ++s) breg[g][s] = bias_d[g * 80 + s * 16 + m];

  v8f cst[5] = {};   // cell state, [tile s][C element]
  v16h Ah[3] = {};   // h(t-1) A-fragments (zero = initial state)
  v8f yacc = {};     // fused FC accumulator (LAYER 1)

  for (int step = 0; step < 48; ++step) {
    const int t = rev ? (47 - step) : step;

    // per-step opaque zero: weight base pointers look loop-variant -> no LICM/spill
    const int oz = opaque_zero();
    const half_t* WihS = Wih_d + oz;
    const half_t* WhhS = sWhh + oz;

    // ---- build input A fragments for time t ----
    v16h Ain[KC];
    if constexpr (LAYER == 0) {
      const float* xr = x + ((size_t)(b0 + m) * 48 + t) * 34;
      if (step + 1 < 48) __builtin_prefetch(x + ((size_t)(b0 + m) * 48 + (rev ? t - 1 : t + 1)) * 34, 0, 3);
#pragma unroll
      for (int kc = 0; kc < 2; ++kc) {
        const int kb = kc * 32 + (hi ? 8 : 0);
        v16h a = {};
#pragma unroll
        for (int e = 0; e < 8; ++e) {
          int k = kb + e;
          if (k < 34) a[e] = (half_t)xr[k];
          int k2 = kb + 16 + e;
          if (k2 < 34) a[e + 8] = (half_t)xr[k2];
        }
        Ain[kc] = a;
      }
    } else {
      const half_t* hr = hin + ((size_t)(b0 + m) * 48 + t) * 136;
      if (step + 1 < 48) __builtin_prefetch(hin + ((size_t)(b0 + m) * 48 + (rev ? t - 1 : t + 1)) * 136, 0, 3);
      v8h z = {};
#pragma unroll
      for (int kc = 0; kc < 5; ++kc) {
        const int kb = kc * 32 + (hi ? 8 : 0);
        V16 u;
        u.h[0] = (kb + 8 <= 136) ? *(const v8h*)(hr + kb) : z;
        u.h[1] = (kb + 24 <= 136) ? *(const v8h*)(hr + kb + 16) : z;
        Ain[kc] = u.v;
      }
    }

    // ---- gate groups: tile s covers hidden cols s*16..s*16+15 for all 4 gates ----
#pragma unroll
    for (int s = 0; s < 5; ++s) {
      v8f acc[4];
#pragma unroll
      for (int g = 0; g < 4; ++g) {
        v8f c0;
#pragma unroll
        for (int e = 0; e < 8; ++e) c0[e] = breg[g][s];
        const int R = g * 80 + s * 16;
        // input contribution: B tiles from L2-resident global (prepacked fp16)
#pragma unroll
        for (int kc = 0; kc < KC; ++kc) {
          const half_t* bp = WihS + (size_t)(R + m) * KIN + kc * 32 + (hi ? 8 : 0);
          V16 bu;
          bu.h[0] = *(const v8h*)bp;
          bu.h[1] = *(const v8h*)(bp + 16);
          c0 = wmma_f16(Ain[kc], bu.v, c0);
        }
        // recurrent contribution: B tiles from LDS
#pragma unroll
        for (int kc = 0; kc < 3; ++kc) {
          const half_t* bp = WhhS + (R + m) * 96 + kc * 32 + (hi ? 8 : 0);
          V16 bu;
          bu.h[0] = *(const v8h*)bp;
          bu.h[1] = *(const v8h*)(bp + 16);
          c0 = wmma_f16(Ah[kc], bu.v, c0);
        }
        acc[g] = c0;
      }
      // elementwise LSTM cell (i,f,g,o aligned per lane/element)
      const int j = s * 16 + m;
#pragma unroll
      for (int e = 0; e < 8; ++e) {
        float ig = fsigm(acc[0][e]);
        float fg = fsigm(acc[1][e]);
        float gg = ftanh(acc[2][e]);
        float og = fsigm(acc[3][e]);
        float c  = fg * cst[s][e] + ig * gg;
        cst[s][e] = c;
        float h  = og * ftanh(c);
        if (j < 68) hT[(e + (hi ? 8 : 0)) * 96 + j] = (half_t)h;  // C-layout -> LDS
      }
    }

    // ---- reload h(t) as A fragments (in-wave DS ordering keeps store->load safe) ----
#pragma unroll
    for (int kc = 0; kc < 3; ++kc) {
      const int kb = kc * 32 + (hi ? 8 : 0);
      V16 u;
      u.h[0] = *(const v8h*)(hT + m * 96 + kb);
      u.h[1] = *(const v8h*)(hT + m * 96 + kb + 16);
      Ah[kc] = u.v;
    }

    if constexpr (LAYER == 0) {
      // write h(t) fp16 to global [B][48][136] at channel offset dir*68 (8B stores)
      half_t* orow = hout + ((size_t)(b0 + m) * 48 + t) * 136 + dir * 68;
#pragma unroll
      for (int kc = 0; kc < 3; ++kc) {
#pragma unroll
        for (int r = 0; r < 2; ++r) {
          const int j0 = kc * 32 + (hi ? 8 : 0) + r * 16;
          Q8 w;
          V16 u; u.v = Ah[kc];
          w.v = u.h[r];
          if (j0 < 68)     *(unsigned long long*)(orow + j0)     = w.q[0];
          if (j0 + 4 < 68) *(unsigned long long*)(orow + j0 + 4) = w.q[1];
        }
      }
    } else {
      // fused FC: y[m][n] += sum_c h1[t][m][c] * fc_w[n][t*136 + dir*68 + c]
      const half_t* fb = fcw + (size_t)m * 9216 + t * 192 + dir * 96 + (hi ? 8 : 0);
#pragma unroll
      for (int kc = 0; kc < 3; ++kc) {
        V16 bu;
        bu.h[0] = *(const v8h*)(fb + kc * 32);
        bu.h[1] = *(const v8h*)(fb + kc * 32 + 16);
        yacc = wmma_f16(Ah[kc], bu.v, yacc);
      }
    }
  }

  if constexpr (LAYER == 1) {
    if (m < 5) {
      float* yp = ypart + (size_t)dir * 16384 * 5;
#pragma unroll
      for (int e = 0; e < 8; ++e)
        yp[(size_t)(b0 + e + (hi ? 8 : 0)) * 5 + m] = yacc[e];
    }
  }
}

// ---------------- launcher ----------------
extern "C" void kernel_launch(void* const* d_in, const int* in_sizes, int n_in,
                              void* d_out, int out_size, void* d_ws, size_t ws_size,
                              hipStream_t stream) {
  (void)in_sizes; (void)n_in; (void)out_size; (void)ws_size;

  const float* x = (const float*)d_in[0];

  char* ws = (char*)d_ws;
  size_t off = 0;
  half_t* h0buf = (half_t*)(ws + off); off += (size_t)16384 * 48 * 136 * 2; // 204 MB
  half_t* W0ih  = (half_t*)(ws + off); off += (size_t)2 * 320 * 64 * 2;
  half_t* W0hh  = (half_t*)(ws + off); off += (size_t)2 * 320 * 96 * 2;
  half_t* W1ih  = (half_t*)(ws + off); off += (size_t)2 * 320 * 160 * 2;
  half_t* W1hh  = (half_t*)(ws + off); off += (size_t)2 * 320 * 96 * 2;
  float*  bias0 = (float*)(ws + off);  off += (size_t)2 * 320 * 4;
  float*  bias1 = (float*)(ws + off);  off += (size_t)2 * 320 * 4;
  half_t* fcwp  = (half_t*)(ws + off); off += (size_t)16 * 48 * 2 * 96 * 2;
  float*  ypart = (float*)(ws + off);  off += (size_t)2 * 16384 * 5 * 4;

  // ---- prepack weights (fp32 -> padded fp16, gate-grouped rows) ----
  auto packW = [&](half_t* dst, const void* src, int sk, int dk) {
    int tot = 320 * dk;
    pack_gates_kernel<<<(tot + 255) / 256, 256, 0, stream>>>(dst, (const float*)src, sk, dk);
  };
  packW(W0ih,             d_in[1],  34, 64);
  packW(W0ih + 320 * 64,  d_in[5],  34, 64);
  packW(W0hh,             d_in[2],  68, 96);
  packW(W0hh + 320 * 96,  d_in[6],  68, 96);
  packW(W1ih,             d_in[9],  136, 160);
  packW(W1ih + 320 * 160, d_in[13], 136, 160);
  packW(W1hh,             d_in[10], 68, 96);
  packW(W1hh + 320 * 96,  d_in[14], 68, 96);

  pack_bias_kernel<<<2, 256, 0, stream>>>(bias0,       (const float*)d_in[3],  (const float*)d_in[4]);
  pack_bias_kernel<<<2, 256, 0, stream>>>(bias0 + 320, (const float*)d_in[7],  (const float*)d_in[8]);
  pack_bias_kernel<<<2, 256, 0, stream>>>(bias1,       (const float*)d_in[11], (const float*)d_in[12]);
  pack_bias_kernel<<<2, 256, 0, stream>>>(bias1 + 320, (const float*)d_in[15], (const float*)d_in[16]);

  pack_fc_kernel<<<(16 * 48 * 2 * 96 + 255) / 256, 256, 0, stream>>>(fcwp, (const float*)d_in[17]);

  // ---- LSTM layers (both directions in one grid) ----
  dim3 grid(256, 2);
  size_t smem = (size_t)320 * 96 * 2 + (size_t)4 * 16 * 96 * 2; // 73728 B
  lstm_kernel<0><<<grid, 128, smem, stream>>>(x, nullptr, W0ih, W0hh, bias0, h0buf, nullptr, nullptr);
  lstm_kernel<1><<<grid, 128, smem, stream>>>(nullptr, h0buf, W1ih, W1hh, bias1, nullptr, fcwp, ypart);

  // ---- combine fwd/bwd FC partials + bias ----
  fc_final_kernel<<<(16384 * 5 + 255) / 256, 256, 0, stream>>>((float*)d_out, ypart,
                                                               (const float*)d_in[18], 16384 * 5);
}
